// ResidualVQEMA_65309272703241
// MI455X (gfx1250) — matrix operand order repaired and checked by reading, not compiled, not run on hardware
//
#include <hip/hip_runtime.h>
#include <hip/hip_bf16.h>

typedef __attribute__((ext_vector_type(16))) __bf16 v16bf;
typedef __attribute__((ext_vector_type(8)))  float  v8f;

#define D_DIM        256
#define T_DIM        4096
#define E_NUM        1024
#define E_HALF       512   // codes scanned per wave (pair splits the codebook)
#define WAVES_PER_WG 8
#define PAIRS_PER_WG 4
#define TOK_PER_PAIR 32    // two 16-row A tiles shared by a wave pair
#define LDS_STRIDE   260   // 256 + 4 pad: (4*m) mod 64 distinct for m=0..15 -> conflict-free

// ---------------------------------------------------------------------------
// Prep: books fp32 -> bf16, and bias[row] = 0.5 * ||e||^2
// ---------------------------------------------------------------------------
__global__ void vq_prep_kernel(const float* __restrict__ books,
                               __bf16* __restrict__ books_bf,
                               float* __restrict__ bias, int n_rows) {
  int row = blockIdx.x;
  if (row >= n_rows) return;
  int t = threadIdx.x;                       // 0..255 == D
  float v = books[(size_t)row * D_DIM + t];
  books_bf[(size_t)row * D_DIM + t] = (__bf16)v;
  float sq = v * v;
  #pragma unroll
  for (int off = 16; off >= 1; off >>= 1) sq += __shfl_down(sq, off, 32);
  __shared__ float partial[8];
  if ((t & 31) == 0) partial[t >> 5] = sq;
  __syncthreads();
  if (t == 0) {
    float s = 0.f;
    #pragma unroll
    for (int i = 0; i < 8; ++i) s += partial[i];
    bias[row] = 0.5f * s;
  }
}

// ---------------------------------------------------------------------------
// Main: a wave PAIR owns 32 tokens (fp32 residual in LDS). Even wave scans
// codes [0,512), odd wave scans [512,1024) -> 2x the waves, same B traffic.
// Scores via v_wmma_f32_16x16x32_bf16, 4 independent accumulator chains,
// each B fragment reused by 2 WMMAs (two M-tiles).
// ---------------------------------------------------------------------------
__global__ __launch_bounds__(256, 1) void vq_main_kernel(
    const float* __restrict__ z, const float* __restrict__ books,
    const __bf16* __restrict__ books_bf, const float* __restrict__ bias,
    const int* __restrict__ n_books_ptr, float* __restrict__ out) {

  __shared__ float res_lds[PAIRS_PER_WG][TOK_PER_PAIR * LDS_STRIDE]; // ~133 KB
  __shared__ float pb_val[PAIRS_PER_WG][2][TOK_PER_PAIR];
  __shared__ int   pb_idx[PAIRS_PER_WG][2][TOK_PER_PAIR];

  const int lane  = threadIdx.x & 31;
  const int wv    = threadIdx.x >> 5;
  const int pr    = wv >> 1;                 // pair within WG
  const int par   = wv & 1;                  // 0: low codes + tile0 duty; 1: high + tile1
  const int gpair = blockIdx.x * PAIRS_PER_WG + pr;
  const int tok0  = gpair * TOK_PER_PAIR;
  const int b     = tok0 / T_DIM;            // T % 32 == 0 -> one batch per pair
  const int t0    = tok0 % T_DIM;

  const int m = lane & 15;                   // row within a 16-token tile
  const int h = lane >> 4;                   // lane half (owns disjoint K-subsets)
  float* myres = res_lds[pr];

  // residual = x = z[b, :, token]; each wave loads its duty tile's rows
  const float* zb = z + (size_t)b * D_DIM * T_DIM + (size_t)(t0 + m);
  {
    const float* zt = zb + 16 * par;
    float* rrow = myres + (m + 16 * par) * LDS_STRIDE;
    #pragma unroll
    for (int kc = 0; kc < 8; ++kc) {
      const int k0 = kc * 32 + 8 * h;
      #pragma unroll
      for (int j = 0; j < 8; ++j) {
        rrow[k0 + j]      = zt[(size_t)(k0 + j) * T_DIM];
        rrow[k0 + 16 + j] = zt[(size_t)(k0 + 16 + j) * T_DIM];
      }
    }
  }
  __syncthreads();

  const int NB    = n_books_ptr[0];
  const int nbase = par * E_HALF;
  for (int bk = 0; bk < NB; ++bk) {
    // ---- A fragments (both tiles, 16x32 bf16 per K-chunk)
    // lane(m,h), elem j -> K = kc*32 + 16h + j
    v16bf a0[8], a1[8];
    #pragma unroll
    for (int kc = 0; kc < 8; ++kc) {
      #pragma unroll
      for (int j = 0; j < 16; ++j) {
        const int k = kc * 32 + 16 * h + j;
        a0[kc][j] = (__bf16)myres[m * LDS_STRIDE + k];
        a1[kc][j] = (__bf16)myres[(m + 16) * LDS_STRIDE + k];
      }
    }

    float bv0[8], bv1[8];
    int   bi0[8], bi1[8];
    #pragma unroll
    for (int r = 0; r < 8; ++r) {
      bv0[r] = -3.4e38f; bi0[r] = nbase;
      bv1[r] = -3.4e38f; bi1[r] = nbase;
    }

    // B layout (32x16): lane n=lane&15 holds code row; elem j -> K = kc*32 + 16h + j
    const __bf16* bb    = books_bf + ((size_t)bk * E_NUM + nbase + (lane & 15)) * D_DIM + 16 * h;
    const float*  biasb = bias + bk * E_NUM + nbase + (lane & 15);

    for (int n0 = 0; n0 < E_HALF; n0 += 32) {  // two 16-code N-tiles per iter
      v8f c00 = {}, c01 = {}, c10 = {}, c11 = {};
      const __bf16* p0 = bb + (size_t)n0 * D_DIM;
      const __bf16* p1 = p0 + (size_t)16 * D_DIM;
      #pragma unroll
      for (int kc = 0; kc < 8; ++kc) {          // each B fragment feeds 2 WMMAs
        v16bf b0 = *(const v16bf*)(p0 + kc * 32);
        v16bf b1 = *(const v16bf*)(p1 + kc * 32);
        c00 = __builtin_amdgcn_wmma_f32_16x16x32_bf16(false, a0[kc], false, b0,
                                                      (short)0, c00, false, false);
        c10 = __builtin_amdgcn_wmma_f32_16x16x32_bf16(false, a1[kc], false, b0,
                                                      (short)0, c10, false, false);
        c01 = __builtin_amdgcn_wmma_f32_16x16x32_bf16(false, a0[kc], false, b1,
                                                      (short)0, c01, false, false);
        c11 = __builtin_amdgcn_wmma_f32_16x16x32_bf16(false, a1[kc], false, b1,
                                                      (short)0, c11, false, false);
      }
      const float bias0 = biasb[n0];
      const float bias1 = biasb[n0 + 16];
      const int   i0    = nbase + n0 + (lane & 15);
      #pragma unroll
      for (int r = 0; r < 8; ++r) {             // C: VGPR r, lane L -> M=r+8h, N=L&15
        const float s00 = c00[r] - bias0;
        const float s01 = c01[r] - bias1;
        const float s10 = c10[r] - bias0;
        const float s11 = c11[r] - bias1;
        if (s00 > bv0[r]) { bv0[r] = s00; bi0[r] = i0; }
        if (s01 > bv0[r]) { bv0[r] = s01; bi0[r] = i0 + 16; }
        if (s10 > bv1[r]) { bv1[r] = s10; bi1[r] = i0; }
        if (s11 > bv1[r]) { bv1[r] = s11; bi1[r] = i0 + 16; }
      }
    }

    // argmax across the 16 lanes of each half (xor masks stay within the half)
    #pragma unroll
    for (int off = 8; off >= 1; off >>= 1) {
      #pragma unroll
      for (int r = 0; r < 8; ++r) {
        const float ov0 = __shfl_xor(bv0[r], off, 32);
        const int   oi0 = __shfl_xor(bi0[r], off, 32);
        if (ov0 > bv0[r] || (ov0 == bv0[r] && oi0 < bi0[r])) { bv0[r] = ov0; bi0[r] = oi0; }
        const float ov1 = __shfl_xor(bv1[r], off, 32);
        const int   oi1 = __shfl_xor(bi1[r], off, 32);
        if (ov1 > bv1[r] || (ov1 == bv1[r] && oi1 < bi1[r])) { bv1[r] = ov1; bi1[r] = oi1; }
      }
    }
    // publish this wave's half-codebook winner per token
    if ((lane & 15) == 0) {
      #pragma unroll
      for (int r = 0; r < 8; ++r) {
        pb_val[pr][par][r + 8 * h]      = bv0[r];
        pb_idx[pr][par][r + 8 * h]      = bi0[r];
        pb_val[pr][par][16 + r + 8 * h] = bv1[r];
        pb_idx[pr][par][16 + r + 8 * h] = bi1[r];
      }
    }
    __syncthreads();

    // merge halves for my duty row; even wave idx < 512 <= odd wave idx,
    // so ties resolve to the lower index (matches jnp.argmax first-match)
    const int   row = m + 16 * par;
    const float va  = pb_val[pr][0][row];
    const int   ia  = pb_idx[pr][0][row];
    const float vb  = pb_val[pr][1][row];
    const int   ib  = pb_idx[pr][1][row];
    const int   idx = (vb > va) ? ib : ia;

    // residual -= books[bk][idx] (exact fp32; h-halves own disjoint K-subsets)
    const float* qrow = books + ((size_t)bk * E_NUM + idx) * D_DIM;
    float* rrow = myres + row * LDS_STRIDE;
    #pragma unroll
    for (int kc = 0; kc < 8; ++kc) {
      const int k0 = kc * 32 + 8 * h;
      const float4 q0 = *(const float4*)(qrow + k0);
      const float4 q1 = *(const float4*)(qrow + k0 + 4);
      const float4 q2 = *(const float4*)(qrow + k0 + 16);
      const float4 q3 = *(const float4*)(qrow + k0 + 20);
      float* rr = rrow + k0;
      rr[0]  -= q0.x; rr[1]  -= q0.y; rr[2]  -= q0.z; rr[3]  -= q0.w;
      rr[4]  -= q1.x; rr[5]  -= q1.y; rr[6]  -= q1.z; rr[7]  -= q1.w;
      rr[16] -= q2.x; rr[17] -= q2.y; rr[18] -= q2.z; rr[19] -= q2.w;
      rr[20] -= q3.x; rr[21] -= q3.y; rr[22] -= q3.z; rr[23] -= q3.w;
    }
    __syncthreads();
  }

  // out = x - residual_final; each wave writes its duty tile, (B, D, T) layout
  {
    const float* zt   = zb + 16 * par;
    float*       ot   = out + (size_t)b * D_DIM * T_DIM + (size_t)(t0 + m) + 16 * par;
    const float* rrow = myres + (m + 16 * par) * LDS_STRIDE;
    #pragma unroll
    for (int kc = 0; kc < 8; ++kc) {
      const int k0 = kc * 32 + 8 * h;
      #pragma unroll
      for (int j = 0; j < 8; ++j) {
        ot[(size_t)(k0 + j) * T_DIM]      = zt[(size_t)(k0 + j) * T_DIM]      - rrow[k0 + j];
        ot[(size_t)(k0 + 16 + j) * T_DIM] = zt[(size_t)(k0 + 16 + j) * T_DIM] - rrow[k0 + 16 + j];
      }
    }
  }
}

// ---------------------------------------------------------------------------
extern "C" void kernel_launch(void* const* d_in, const int* in_sizes, int n_in,
                              void* d_out, int out_size, void* d_ws, size_t ws_size,
                              hipStream_t stream) {
  const float* z       = (const float*)d_in[0];   // (16, 256, 4096) fp32
  const float* books   = (const float*)d_in[1];   // (8, 1024, 256) fp32
  const int*   n_books = (const int*)d_in[2];     // scalar
  float*       out     = (float*)d_out;           // (16, 256, 4096) fp32

  const size_t books_elems = (size_t)in_sizes[1];                  // 8*1024*256
  __bf16* books_bf = (__bf16*)d_ws;
  float*  bias     = (float*)((char*)d_ws + books_elems * sizeof(__bf16));
  const int n_rows = (int)(books_elems / D_DIM);                   // 8192

  vq_prep_kernel<<<n_rows, 256, 0, stream>>>(books, books_bf, bias, n_rows);

  const int total_tokens = in_sizes[0] / D_DIM;                       // 65536
  const int n_blocks = total_tokens / (PAIRS_PER_WG * TOK_PER_PAIR);  // 512
  vq_main_kernel<<<n_blocks, 256, 0, stream>>>(z, books, books_bf, bias, n_books, out);
}
